// NeuronGAT_40535901339860
// MI455X (gfx1250) — matrix operand order repaired and checked by reading, not compiled, not run on hardware
//
#include <hip/hip_runtime.h>
#include <hip/hip_bf16.h>

// ---------------- problem constants (match reference) ----------------
#define N_NODES   50000
#define N_EDGES   800000
#define ET_EDGES  (N_EDGES + N_NODES)   // with self loops
#define IN_F      128
#define HID_C     32
#define HEADS     4
#define D1        128                   // HEADS*HID
#define OUT_F     10
#define NEG_SLOPE 0.2f
#define BN_EPS    1e-5f

typedef _Float16 half8   __attribute__((ext_vector_type(8)));
typedef _Float16 half16  __attribute__((ext_vector_type(16)));
typedef float    floatx8 __attribute__((ext_vector_type(8)));

// monotone float<->int encoding so atomicMax(int) == max over floats
__device__ __forceinline__ int float_enc(float f) {
    int i = __float_as_int(f);
    return i ^ ((i >> 31) & 0x7fffffff);
}
__device__ __forceinline__ float float_dec(int i) {
    return __int_as_float(i ^ ((i >> 31) & 0x7fffffff));
}

__device__ __forceinline__ void get_edge(const int* __restrict__ ei, int i, int& s, int& d) {
    if (i < N_EDGES) { s = ei[i]; d = ei[N_EDGES + i]; }
    else             { s = i - N_EDGES; d = s; }
}

// ---------------- utility kernels ----------------
__global__ void fill_f32_kernel(float* p, float v, long long n) {
    long long t = blockIdx.x * (long long)blockDim.x + threadIdx.x;
    if (t < n) p[t] = v;
}
__global__ void fill_i32_kernel(int* p, int v, long long n) {
    long long t = blockIdx.x * (long long)blockDim.x + threadIdx.x;
    if (t < n) p[t] = v;
}
__global__ void cvt_f32_to_f16_kernel(const float* __restrict__ in, _Float16* __restrict__ out, long long n) {
    long long t = blockIdx.x * (long long)blockDim.x + threadIdx.x;
    if (t < n) out[t] = (_Float16)in[t];
}

// ---------------- WMMA GEMM: C[M, NT*16] = A[M,K](f16) @ W[NT*16, K](f16)^T ---------
// One wave per 16-row tile; NT column tiles kept in v8f accumulators.
template<int NT, int KDIM>
__global__ void wmma_gemm_kernel(const _Float16* __restrict__ A,
                                 const _Float16* __restrict__ W,
                                 float* __restrict__ C, int M) {
    const int wave = threadIdx.x >> 5;
    const int lane = threadIdx.x & 31;
    const int tile = blockIdx.x * (blockDim.x >> 5) + wave;
    if (tile * 16 >= M) return;              // uniform per-wave: EXEC stays all-ones

    const int nlo  = lane & 15;
    const int halF = lane >> 4;

    floatx8 acc[NT];
    #pragma unroll
    for (int i = 0; i < NT; i++) acc[i] = (floatx8){0.f,0.f,0.f,0.f,0.f,0.f,0.f,0.f};

    const long long rowA = (long long)(tile * 16 + nlo) * KDIM;

    #pragma unroll
    for (int kb = 0; kb < KDIM; kb += 32) {
        // A fragment (16x32 f16): lanes 0-15 hold K kb+{0..7,16..23}, lanes 16-31 K kb+{8..15,24..31}
        half8 alo = *(const half8*)(A + rowA + kb +      8 * halF);
        half8 ahi = *(const half8*)(A + rowA + kb + 16 + 8 * halF);
        half16 av;
        #pragma unroll
        for (int i = 0; i < 8; i++) { av[i] = alo[i]; av[i + 8] = ahi[i]; }

        #pragma unroll
        for (int nt = 0; nt < NT; nt++) {
            // B fragment (32x16 f16): lane = column n, lanes 0-15 hold K kb+{0..7,16..23}, lanes 16-31 K kb+{8..15,24..31}
            const _Float16* wp = W + (long long)(nt * 16 + nlo) * KDIM + kb + 8 * halF;
            half8 blo = *(const half8*)(wp);
            half8 bhi = *(const half8*)(wp + 16);
            half16 bv;
            #pragma unroll
            for (int i = 0; i < 8; i++) { bv[i] = blo[i]; bv[i + 8] = bhi[i]; }

            acc[nt] = __builtin_amdgcn_wmma_f32_16x16x32_f16(
                false, av, false, bv, (short)0, acc[nt], false, false);
        }
    }

    // C/D layout: VGPR r, lanes 0-15 -> row r, lanes 16-31 -> row r+8; col = lane&15
    #pragma unroll
    for (int nt = 0; nt < NT; nt++)
        #pragma unroll
        for (int r = 0; r < 8; r++)
            C[(long long)(tile * 16 + r + 8 * halF) * (NT * 16) + nt * 16 + nlo] = acc[nt][r];
}

// ---------------- attention score per (node, head) ----------------
__global__ void score_kernel(const float* __restrict__ feat,
                             const float* __restrict__ a_src,
                             const float* __restrict__ a_dst,
                             float* __restrict__ ssrc, float* __restrict__ sdst,
                             int H, int C, long long total) {
    long long t = blockIdx.x * (long long)blockDim.x + threadIdx.x;
    if (t >= total) return;
    int n = (int)(t / H), h = (int)(t % H);
    const float* fp = feat + (long long)n * H * C + (long long)h * C;
    float s1 = 0.f, s2 = 0.f;
    for (int c = 0; c < C; c++) {
        float v = fp[c];
        s1 += v * a_src[h * C + c];
        s2 += v * a_dst[h * C + c];
    }
    ssrc[t] = s1; sdst[t] = s2;
}

// ---------------- edge pass 1: segment max (int-encoded float atomicMax) ------------
__global__ void edge_max_kernel(const int* __restrict__ ei,
                                const float* __restrict__ ssrc,
                                const float* __restrict__ sdst,
                                int* __restrict__ mbuf, int H, long long total) {
    long long t = blockIdx.x * (long long)blockDim.x + threadIdx.x;
    if (t >= total) return;
    int i = (int)(t / H), h = (int)(t % H);
    int s, d; get_edge(ei, i, s, d);
    float e = ssrc[(long long)s * H + h] + sdst[(long long)d * H + h];
    e = e > 0.f ? e : NEG_SLOPE * e;
    atomicMax(&mbuf[(long long)d * H + h], float_enc(e));
}

// ---------------- edge pass 2: softmax denominator ----------------
__global__ void edge_expsum_kernel(const int* __restrict__ ei,
                                   const float* __restrict__ ssrc,
                                   const float* __restrict__ sdst,
                                   const int* __restrict__ mbuf,
                                   float* __restrict__ denom, int H, long long total) {
    long long t = blockIdx.x * (long long)blockDim.x + threadIdx.x;
    if (t >= total) return;
    int i = (int)(t / H), h = (int)(t % H);
    int s, d; get_edge(ei, i, s, d);
    float e = ssrc[(long long)s * H + h] + sdst[(long long)d * H + h];
    e = e > 0.f ? e : NEG_SLOPE * e;
    float w = __expf(e - float_dec(mbuf[(long long)d * H + h]));
    __hip_atomic_fetch_add(&denom[(long long)d * H + h], w,
                           __ATOMIC_RELAXED, __HIP_MEMORY_SCOPE_AGENT);
}

// ---------------- edge pass 3: normalized message scatter-add ----------------
// thread = (edge, channel); consecutive channels -> coalesced gather + coalesced atomics
__global__ void edge_agg_kernel(const int* __restrict__ ei,
                                const float* __restrict__ ssrc,
                                const float* __restrict__ sdst,
                                const int* __restrict__ mbuf,
                                const float* __restrict__ denom,
                                const float* __restrict__ feat,
                                float* __restrict__ outAgg,
                                int H, int C, long long total) {
    long long t = blockIdx.x * (long long)blockDim.x + threadIdx.x;
    if (t >= total) return;
    const int F = H * C;
    int i = (int)(t / F), c = (int)(t % F), h = c / C;
    int s, d; get_edge(ei, i, s, d);
    float e = ssrc[(long long)s * H + h] + sdst[(long long)d * H + h];
    e = e > 0.f ? e : NEG_SLOPE * e;
    float alpha = __expf(e - float_dec(mbuf[(long long)d * H + h])) /
                  denom[(long long)d * H + h];
    float msg = alpha * feat[(long long)s * F + c];
    __hip_atomic_fetch_add(&outAgg[(long long)d * F + c], msg,
                           __ATOMIC_RELAXED, __HIP_MEMORY_SCOPE_AGENT);
}

// ---------------- bias + BN(eval) + ELU; optional f16 and/or f32 outputs ------------
__global__ void bn_elu_kernel(const float* __restrict__ agg,
                              const float* __restrict__ bias,
                              const float* __restrict__ gamma,
                              const float* __restrict__ beta,
                              const float* __restrict__ mean,
                              const float* __restrict__ var,
                              _Float16* __restrict__ outh,
                              float* __restrict__ outf,
                              int F, long long total) {
    long long t = blockIdx.x * (long long)blockDim.x + threadIdx.x;
    if (t >= total) return;
    int c = (int)(t % F);
    float v = agg[t] + bias[c];
    v = (v - mean[c]) * (gamma[c] * rsqrtf(var[c] + BN_EPS)) + beta[c];
    v = v > 0.f ? v : expm1f(v);          // ELU, alpha=1
    if (outh) outh[t] = (_Float16)v;
    if (outf) outf[t] = v;
}

// ---------------- final classifier: out[N,10] = h2bn[N,32] @ Wc^T + bc --------------
__global__ void classify_kernel(const float* __restrict__ feat,
                                const float* __restrict__ Wc,
                                const float* __restrict__ bc,
                                float* __restrict__ out, long long total) {
    long long t = blockIdx.x * (long long)blockDim.x + threadIdx.x;
    if (t >= total) return;
    int n = (int)(t / OUT_F), o = (int)(t % OUT_F);
    float s = bc[o];
    const float* fp = feat + (long long)n * HID_C;
    for (int c = 0; c < HID_C; c++) s += fp[c] * Wc[o * HID_C + c];
    out[t] = s;
}

// ==================================================================================
extern "C" void kernel_launch(void* const* d_in, const int* in_sizes, int n_in,
                              void* d_out, int out_size, void* d_ws, size_t ws_size,
                              hipStream_t stream) {
    const float* x        = (const float*)d_in[0];
    const int*   ei       = (const int*)  d_in[1];
    const float* W1       = (const float*)d_in[2];
    const float* a_src1   = (const float*)d_in[3];
    const float* a_dst1   = (const float*)d_in[4];
    const float* b1       = (const float*)d_in[5];
    const float* bn1_g    = (const float*)d_in[6];
    const float* bn1_b    = (const float*)d_in[7];
    const float* bn1_m    = (const float*)d_in[8];
    const float* bn1_v    = (const float*)d_in[9];
    const float* W2       = (const float*)d_in[10];
    const float* a_src2   = (const float*)d_in[11];
    const float* a_dst2   = (const float*)d_in[12];
    const float* b2       = (const float*)d_in[13];
    const float* bn2_g    = (const float*)d_in[14];
    const float* bn2_b    = (const float*)d_in[15];
    const float* bn2_m    = (const float*)d_in[16];
    const float* bn2_v    = (const float*)d_in[17];
    const float* Wc       = (const float*)d_in[18];
    const float* bc       = (const float*)d_in[19];
    float* out = (float*)d_out;

    // ---- workspace carve-up (256B aligned regions) ----
    char* ws = (char*)d_ws;
    size_t off = 0;
    auto carve = [&](size_t bytes) -> char* {
        char* p = ws + off;
        off = (off + bytes + 255) & ~(size_t)255;
        return p;
    };
    _Float16* xh    = (_Float16*)carve((size_t)N_NODES * D1 * 2);   // later reused as h1bn f16
    _Float16* W1h   = (_Float16*)carve((size_t)D1 * IN_F * 2);
    _Float16* W2h   = (_Float16*)carve((size_t)HID_C * D1 * 2);
    float*    h1    = (float*)   carve((size_t)N_NODES * D1 * 4);   // start reused as out2
    float*    out1  = (float*)   carve((size_t)N_NODES * D1 * 4);   // later reused as h2bn
    float*    ssrc1 = (float*)   carve((size_t)N_NODES * HEADS * 4);
    float*    sdst1 = (float*)   carve((size_t)N_NODES * HEADS * 4);
    int*      m1    = (int*)     carve((size_t)N_NODES * HEADS * 4);
    float*    den1  = (float*)   carve((size_t)N_NODES * HEADS * 4);
    float*    h2    = (float*)   carve((size_t)N_NODES * HID_C * 4);
    float*    ssrc2 = (float*)   carve((size_t)N_NODES * 4);
    float*    sdst2 = (float*)   carve((size_t)N_NODES * 4);
    int*      m2    = (int*)     carve((size_t)N_NODES * 4);
    float*    den2  = (float*)   carve((size_t)N_NODES * 4);
    // aliases (lifetimes do not overlap):
    _Float16* h1bn_h = xh;      // layer-2 GEMM A operand (f16), written after gemm1 consumed xh
    float*    out2   = h1;      // layer-2 aggregate, written after h1 consumed by edge_agg1
    float*    h2bn   = out1;    // layer-2 BN output, written after out1 consumed by bn1

    const int B = 256;
    auto NB = [](long long total, int blk) -> unsigned {
        return (unsigned)((total + blk - 1) / blk);
    };

    // ---- 0) f32 -> f16 conversions ----
    cvt_f32_to_f16_kernel<<<NB((long long)N_NODES * IN_F, B), B, 0, stream>>>(x,  xh,  (long long)N_NODES * IN_F);
    cvt_f32_to_f16_kernel<<<NB((long long)D1 * IN_F,      B), B, 0, stream>>>(W1, W1h, (long long)D1 * IN_F);
    cvt_f32_to_f16_kernel<<<NB((long long)HID_C * D1,     B), B, 0, stream>>>(W2, W2h, (long long)HID_C * D1);

    // ---- 1) GEMM1: h1[N,128] = x @ W1^T  (WMMA f16->f32) ----
    {
        int tiles = N_NODES / 16;                       // 3125
        int blocks = (tiles + 7) / 8;                   // 8 waves/block
        wmma_gemm_kernel<8, 128><<<blocks, 256, 0, stream>>>(xh, W1h, h1, N_NODES);
    }

    // ---- 2) per-node attention scores (layer 1) ----
    score_kernel<<<NB((long long)N_NODES * HEADS, B), B, 0, stream>>>(
        h1, a_src1, a_dst1, ssrc1, sdst1, HEADS, HID_C, (long long)N_NODES * HEADS);

    // ---- 3) init accumulators ----
    fill_i32_kernel<<<NB((long long)N_NODES * HEADS, B), B, 0, stream>>>(m1, INT_MIN, (long long)N_NODES * HEADS);
    fill_f32_kernel<<<NB((long long)N_NODES * HEADS, B), B, 0, stream>>>(den1, 0.f, (long long)N_NODES * HEADS);
    fill_f32_kernel<<<NB((long long)N_NODES * D1,    B), B, 0, stream>>>(out1, 0.f, (long long)N_NODES * D1);

    // ---- 4) edge passes (layer 1) ----
    edge_max_kernel<<<NB((long long)ET_EDGES * HEADS, B), B, 0, stream>>>(
        ei, ssrc1, sdst1, m1, HEADS, (long long)ET_EDGES * HEADS);
    edge_expsum_kernel<<<NB((long long)ET_EDGES * HEADS, B), B, 0, stream>>>(
        ei, ssrc1, sdst1, m1, den1, HEADS, (long long)ET_EDGES * HEADS);
    edge_agg_kernel<<<NB((long long)ET_EDGES * D1, B), B, 0, stream>>>(
        ei, ssrc1, sdst1, m1, den1, h1, out1, HEADS, HID_C, (long long)ET_EDGES * D1);

    // ---- 5) bias + BN + ELU -> f16 features for layer 2 ----
    bn_elu_kernel<<<NB((long long)N_NODES * D1, B), B, 0, stream>>>(
        out1, b1, bn1_g, bn1_b, bn1_m, bn1_v, h1bn_h, (float*)nullptr,
        D1, (long long)N_NODES * D1);

    // ---- 6) GEMM2: h2[N,32] = h1bn @ W2^T ----
    {
        int tiles = N_NODES / 16;
        int blocks = (tiles + 7) / 8;
        wmma_gemm_kernel<2, 128><<<blocks, 256, 0, stream>>>(h1bn_h, W2h, h2, N_NODES);
    }

    // ---- 7) scores (layer 2, H=1) ----
    score_kernel<<<NB((long long)N_NODES, B), B, 0, stream>>>(
        h2, a_src2, a_dst2, ssrc2, sdst2, 1, HID_C, (long long)N_NODES);

    // ---- 8) init layer-2 accumulators ----
    fill_i32_kernel<<<NB((long long)N_NODES, B), B, 0, stream>>>(m2, INT_MIN, (long long)N_NODES);
    fill_f32_kernel<<<NB((long long)N_NODES, B), B, 0, stream>>>(den2, 0.f, (long long)N_NODES);
    fill_f32_kernel<<<NB((long long)N_NODES * HID_C, B), B, 0, stream>>>(out2, 0.f, (long long)N_NODES * HID_C);

    // ---- 9) edge passes (layer 2) ----
    edge_max_kernel<<<NB((long long)ET_EDGES, B), B, 0, stream>>>(
        ei, ssrc2, sdst2, m2, 1, (long long)ET_EDGES);
    edge_expsum_kernel<<<NB((long long)ET_EDGES, B), B, 0, stream>>>(
        ei, ssrc2, sdst2, m2, den2, 1, (long long)ET_EDGES);
    edge_agg_kernel<<<NB((long long)ET_EDGES * HID_C, B), B, 0, stream>>>(
        ei, ssrc2, sdst2, m2, den2, h2, out2, 1, HID_C, (long long)ET_EDGES * HID_C);

    // ---- 10) bias + BN + ELU (f32 out) ----
    bn_elu_kernel<<<NB((long long)N_NODES * HID_C, B), B, 0, stream>>>(
        out2, b2, bn2_g, bn2_b, bn2_m, bn2_v, (_Float16*)nullptr, h2bn,
        HID_C, (long long)N_NODES * HID_C);

    // ---- 11) classifier ----
    classify_kernel<<<NB((long long)N_NODES * OUT_F, B), B, 0, stream>>>(
        h2bn, Wc, bc, out, (long long)N_NODES * OUT_F);
}
